// DotProductAttention_69320772157489
// MI455X (gfx1250) — compile-verified, matrix operand order checked by reading, not run on hardware
//
#include <hip/hip_runtime.h>
#include <stdint.h>
#include <stddef.h>

// ---------------------------------------------------------------------------
// CDNA5 (gfx1250) types & helpers
// ---------------------------------------------------------------------------
typedef __bf16 bf16;
typedef __attribute__((ext_vector_type(16))) __bf16 v16bf;
typedef __attribute__((ext_vector_type(8)))  __bf16 v8bf;
typedef __attribute__((ext_vector_type(8)))  float  v8f;

#define DEV __device__ __forceinline__

// Problem dims (compile-time)
constexpr int kB   = 1024;
constexpr int kS   = 64;
constexpr int kH   = 16;
constexpr int kDK  = 64;
constexpr int kD   = 1024;   // H*DK
constexpr int kSMSQ = 32;
constexpr int kSMI  = 256;

DEV bf16 cvt_bf16(float f) {
  union { float f; uint32_t u; } c; c.f = f;
  uint32_t r = c.u + 0x7FFFu + ((c.u >> 16) & 1u);   // round-to-nearest-even
  uint16_t h = (uint16_t)(r >> 16);
  bf16 o; __builtin_memcpy(&o, &h, 2);
  return o;
}

DEV v8f vzero() {
  v8f z;
  #pragma unroll
  for (int i = 0; i < 8; ++i) z[i] = 0.f;
  return z;
}

DEV v8f wmma_bf16(v16bf a, v16bf b, v8f c) {
  // D = A(16x32 bf16) * B(32x16 bf16) + C(16x16 f32)
  return __builtin_amdgcn_wmma_f32_16x16x32_bf16(
      /*neg_a=*/false, a, /*neg_b=*/false, b,
      /*c_mod=*/(short)0, c, /*reuse_a=*/false, /*reuse_b=*/false);
}

// Fragment loader matching the CDNA5 16-bit operand layout:
//   row (M or N) = row0 + (lane & 15)
//   K groups: lanes 0-15 hold K={k0..k0+7, k0+16..k0+23},
//             lanes 16-31 hold K={k0+8..k0+15, k0+24..k0+31}
// Storage is row-major [row][ld]; works for A ([m][k]) and for B staged as
// [n][k] (i.e. B-transposed storage, which is the natural torch weight layout).
DEV v16bf load_frag(const bf16* base, int row0, int ld, int k0, int lane) {
  const bf16* p = base + (size_t)(row0 + (lane & 15)) * ld + k0 + ((lane >> 4) << 3);
  union { v16bf v; v8bf h[2]; } u;
  u.h[0] = *(const v8bf*)(p);
  u.h[1] = *(const v8bf*)(p + 16);
  return u.v;
}

// ---------------------------------------------------------------------------
// Stage 0: f32 -> bf16 pack (activations + matmul weights)
// ---------------------------------------------------------------------------
__global__ __launch_bounds__(256) void k_pack_bf16(const float* __restrict__ src,
                                                   bf16* __restrict__ dst, long n4) {
  long i = (long)blockIdx.x * blockDim.x + threadIdx.x;
  if (i >= n4) return;
  const float4 v = ((const float4*)src)[i];
  bf16 o[4] = { cvt_bf16(v.x), cvt_bf16(v.y), cvt_bf16(v.z), cvt_bf16(v.w) };
  uint64_t pk; __builtin_memcpy(&pk, o, 8);
  ((uint64_t*)dst)[i] = pk;
}

// ---------------------------------------------------------------------------
// Stages 1 & 6 core: double-buffered 64xK @ KxN(=256-wide block) bf16 GEMM.
// 8 waves; wave -> (tile_m, 8 n-tiles): A-frag reuse across 8 back-to-back
// WMMAs; all 16 ds_load_b128 issued first so the compiler clauses them and
// the WMMA chain runs without per-op dscnt stalls.  One barrier per K chunk.
// ---------------------------------------------------------------------------
struct GemmCtx {
  const bf16* Ab;   // [64][kD] row-major activations
  const bf16* Wb;   // [N][kD] torch-layout weights (rows = out channel)
  int n0;           // N-block base
  int tid, lane, wave, tile_m, tnb;
};

DEV void gemm_64xK_256(GemmCtx g, bf16 (*As)[64 * 32], bf16 (*Bs)[256 * 32],
                       v8f acc[8]) {
  const int ra = g.tid >> 2, ca = (g.tid & 3) * 8;
  const bf16* wrow = g.Wb + (size_t)(g.n0 + g.tid) * kD;

  // prologue: stage chunk 0 into buffer 0
  *(v8bf*)&As[0][ra * 32 + ca] = *(const v8bf*)&g.Ab[(size_t)ra * kD + ca];
  #pragma unroll
  for (int j = 0; j < 4; ++j)
    *(v8bf*)&Bs[0][g.tid * 32 + j * 8] = *(const v8bf*)(wrow + j * 8);

  int p = 0;
  for (int kc = 0; kc < kD; kc += 32) {
    __syncthreads();                        // buffer p staged, buffer p^1 free
    if (kc + 32 < kD) {                     // stage next chunk while computing
      *(v8bf*)&As[p ^ 1][ra * 32 + ca] =
          *(const v8bf*)&g.Ab[(size_t)ra * kD + kc + 32 + ca];
      #pragma unroll
      for (int j = 0; j < 4; ++j)
        *(v8bf*)&Bs[p ^ 1][g.tid * 32 + j * 8] =
            *(const v8bf*)(wrow + kc + 32 + j * 8);
      __builtin_prefetch(wrow + kc + 64, 0, 1);   // global_prefetch_b8
    }
    v16bf a = load_frag(As[p], g.tile_m, 32, 0, g.lane);
    v16bf bb[8];
    #pragma unroll
    for (int t = 0; t < 8; ++t)
      bb[t] = load_frag(Bs[p], (g.tnb + t) * 16, 32, 0, g.lane);
    #pragma unroll
    for (int t = 0; t < 8; ++t)
      acc[t] = wmma_bf16(a, bb[t], acc[t]);
    p ^= 1;
  }
}

// ---------------------------------------------------------------------------
// Stage 1: QKV GEMM.  C[64 x 3072] = query_bf[b] @ qkv_w^T, scattered to
// head-major Q/K/V bf16 tensors [b][h][s][dk].
// ---------------------------------------------------------------------------
__global__ __launch_bounds__(256) void k_qkv_gemm(
    const bf16* __restrict__ qbf, const bf16* __restrict__ wbf,
    bf16* __restrict__ Qo, bf16* __restrict__ Ko, bf16* __restrict__ Vo) {
  __shared__ __align__(16) bf16 As[2][64 * 32];
  __shared__ __align__(16) bf16 Bs[2][256 * 32];
  const int b = blockIdx.x;
  GemmCtx g;
  g.tid = threadIdx.x; g.lane = g.tid & 31; g.wave = g.tid >> 5;
  g.n0 = blockIdx.y * 256;
  g.Ab = qbf + (size_t)b * kS * kD;
  g.Wb = wbf;
  g.tile_m = (g.wave & 3) * 16;
  g.tnb    = (g.wave >> 2) * 8;

  v8f acc[8];
  #pragma unroll
  for (int t = 0; t < 8; ++t) acc[t] = vzero();
  gemm_64xK_256(g, As, Bs, acc);

  // Epilogue: C layout -> element i of lane l is (M = i + 8*(l>>4), N = l&15)
  const int col_l = g.lane & 15, rsel = (g.lane >> 4) * 8;
  #pragma unroll
  for (int t = 0; t < 8; ++t) {
    const int n  = g.n0 + (g.tnb + t) * 16 + col_l;   // 0..3071
    const int h  = n / 192, r = n % 192, part = r >> 6, dk = r & 63;
    bf16* dst = (part == 0) ? Qo : (part == 1) ? Ko : Vo;
    const size_t basep = (((size_t)b * kH + h) * kS) * kDK + dk;
    #pragma unroll
    for (int i = 0; i < 8; ++i) {
      const int srow = g.tile_m + i + rsel;
      dst[basep + (size_t)srow * kDK] = cvt_bf16(acc[t][i]);
    }
  }
}

// ---------------------------------------------------------------------------
// Stage 6: output projection.  out[b] = hcat[b] @ wh_w^T + wh_b  (f32)
// ---------------------------------------------------------------------------
__global__ __launch_bounds__(256) void k_out_gemm(
    const bf16* __restrict__ hc, const bf16* __restrict__ wbf,
    const float* __restrict__ bias, float* __restrict__ out) {
  __shared__ __align__(16) bf16 As[2][64 * 32];
  __shared__ __align__(16) bf16 Bs[2][256 * 32];
  const int b = blockIdx.x;
  GemmCtx g;
  g.tid = threadIdx.x; g.lane = g.tid & 31; g.wave = g.tid >> 5;
  g.n0 = blockIdx.y * 256;
  g.Ab = hc + (size_t)b * kS * kD;
  g.Wb = wbf;
  g.tile_m = (g.wave & 3) * 16;
  g.tnb    = (g.wave >> 2) * 8;

  v8f acc[8];
  #pragma unroll
  for (int t = 0; t < 8; ++t) acc[t] = vzero();
  gemm_64xK_256(g, As, Bs, acc);

  const int col_l = g.lane & 15, rsel = (g.lane >> 4) * 8;
  #pragma unroll
  for (int t = 0; t < 8; ++t) {
    const int n = g.n0 + (g.tnb + t) * 16 + col_l;
    const float bv = bias[n];
    #pragma unroll
    for (int i = 0; i < 8; ++i) {
      const int srow = g.tile_m + i + rsel;
      out[((size_t)b * kS + srow) * kD + n] = acc[t][i] + bv;
    }
  }
}

// ---------------------------------------------------------------------------
// Stage 2: smolgen layer 1.  s1[b][64*32] = x[b] @ sm1_w^T + sm1_b  (f32 out)
// ---------------------------------------------------------------------------
__global__ __launch_bounds__(256) void k_sm1_gemm(
    const bf16* __restrict__ xbf, const bf16* __restrict__ wbf,
    const float* __restrict__ bias, float* __restrict__ s1) {
  __shared__ __align__(16) bf16 As[64 * 32];
  __shared__ __align__(16) bf16 Bs[32 * 32];
  const int b = blockIdx.x, tid = threadIdx.x, lane = tid & 31, wave = tid >> 5;
  const bf16* Ab = xbf + (size_t)b * kS * kD;
  const int tm = (wave >> 1) * 16, tn = (wave & 1) * 16;

  v8f acc = vzero();
  for (int kc = 0; kc < kD; kc += 32) {
    { int r = tid >> 2, c = (tid & 3) * 8;
      *(v8bf*)&As[r * 32 + c] = *(const v8bf*)&Ab[(size_t)r * kD + kc + c]; }
    if (tid < 128) {
      int r = tid >> 2, c = (tid & 3) * 8;
      *(v8bf*)&Bs[r * 32 + c] = *(const v8bf*)&wbf[(size_t)r * kD + kc + c];
    }
    __syncthreads();
    acc = wmma_bf16(load_frag(As, tm, 32, 0, lane),
                    load_frag(Bs, tn, 32, 0, lane), acc);
    __syncthreads();
  }
  const int n = tn + (lane & 15);
  const float bv = bias[n];
  #pragma unroll
  for (int i = 0; i < 8; ++i) {
    const int srow = tm + i + ((lane >> 4) * 8);
    s1[(size_t)b * (kS * kSMSQ) + srow * kSMSQ + n] = acc[i] + bv;
  }
}

// ---------------------------------------------------------------------------
// Stage 3: smolgen MLP (exact f32): s2 = LN(s1 @ sm2_w^T + b),
// s3 = LN(s2 @ sm3_w^T + b) -> bf16 [b][16][256]  (A operand for prep WMMA)
// ---------------------------------------------------------------------------
DEV float block_sum(float v, float* red, int tid) {
  red[tid] = v; __syncthreads();
  #pragma unroll
  for (int s = 128; s > 0; s >>= 1) {
    if (tid < s) red[tid] += red[tid + s];
    __syncthreads();
  }
  float r = red[0]; __syncthreads();
  return r;
}

__global__ __launch_bounds__(256) void k_smolgen_mlp(
    const float* __restrict__ s1,
    const float* __restrict__ sm2_w, const float* __restrict__ sm2_b,
    const float* __restrict__ ln1_g, const float* __restrict__ ln1_b,
    const float* __restrict__ sm3_w, const float* __restrict__ sm3_b,
    const float* __restrict__ ln2_g, const float* __restrict__ ln2_b,
    bf16* __restrict__ s3bf) {
  __shared__ float ls1[kS * kSMSQ];   // 2048
  __shared__ float ls2[kSMI];         // 256
  __shared__ float red[256];
  const int b = blockIdx.x, tid = threadIdx.x;

  const float* s1b = s1 + (size_t)b * 2048;
  for (int i = tid; i < 2048; i += 256) ls1[i] = s1b[i];
  __syncthreads();

  // s2[tid] = <ls1, sm2_w[tid]> + sm2_b[tid]
  float a = 0.f;
  { const float* wr = sm2_w + (size_t)tid * 2048;
    #pragma unroll 4
    for (int k = 0; k < 2048; ++k) a = fmaf(ls1[k], wr[k], a); }
  a += sm2_b[tid];

  // LayerNorm over 256
  float mu  = block_sum(a, red, tid) * (1.f / 256.f);
  float dv  = a - mu;
  float var = block_sum(dv * dv, red, tid) * (1.f / 256.f);
  ls2[tid]  = dv * rsqrtf(var + 1e-5f) * ln1_g[tid] + ln1_b[tid];
  __syncthreads();

  // s3: 4096 outputs, 16 per thread (out index = j*256 + tid)
  float o[16]; float lsum = 0.f, lsq = 0.f;
  #pragma unroll
  for (int j = 0; j < 16; ++j) {
    const int idx = j * 256 + tid;
    const float* wr = sm3_w + (size_t)idx * kSMI;
    float acc = 0.f;
    #pragma unroll 4
    for (int k = 0; k < kSMI; ++k) acc = fmaf(ls2[k], wr[k], acc);
    acc += sm3_b[idx];
    o[j] = acc; lsum += acc;
  }
  float mu2 = block_sum(lsum, red, tid) * (1.f / 4096.f);
  #pragma unroll
  for (int j = 0; j < 16; ++j) { float d = o[j] - mu2; lsq += d * d; }
  float var2 = block_sum(lsq, red, tid) * (1.f / 4096.f);
  float rs = rsqrtf(var2 + 1e-5f);
  #pragma unroll
  for (int j = 0; j < 16; ++j) {
    const int idx = j * 256 + tid;
    float v = (o[j] - mu2) * rs * ln2_g[idx] + ln2_b[idx];
    s3bf[(size_t)b * 4096 + idx] = cvt_bf16(v);
  }
}

// ---------------------------------------------------------------------------
// Stage 4: smolgen prep.  smol[b][h][q][k] = s3[b][h] @ prep_w^T + prep_b.
// M = 16 heads = exactly one WMMA M tile; A (16x256) fully in LDS,
// B fragments gathered straight from L2-resident prep_w (2 MB).
// ---------------------------------------------------------------------------
__global__ __launch_bounds__(256) void k_smol_prep(
    const bf16* __restrict__ s3bf, const bf16* __restrict__ pw,
    const float* __restrict__ pb, float* __restrict__ smol) {
  __shared__ __align__(16) bf16 As[16 * kSMI];   // 8 KB
  const int b = blockIdx.x, tid = threadIdx.x, lane = tid & 31, wave = tid >> 5;

  for (int i = tid; i < (16 * kSMI) / 8; i += 256)
    ((v8bf*)As)[i] = ((const v8bf*)(s3bf + (size_t)b * 4096))[i];
  __syncthreads();

  for (int j = 0; j < 32; ++j) {
    const int nt = j * 8 + wave;        // n-tile 0..255
    const int n0 = nt * 16;
    v8f acc = vzero();
    #pragma unroll
    for (int kc = 0; kc < kSMI; kc += 32) {
      v16bf a  = load_frag(As, 0, kSMI, kc, lane);
      v16bf bb = load_frag(pw + (size_t)n0 * kSMI, 0, kSMI, kc, lane);
      acc = wmma_bf16(a, bb, acc);
    }
    const int col = n0 + (lane & 15);           // 0..4095 => (q, k)
    const int q = col >> 6, kk = col & 63;
    const float bias = pb[col];
    #pragma unroll
    for (int i = 0; i < 8; ++i) {
      const int h = i + ((lane >> 4) * 8);      // 0..15
      smol[((((size_t)b * kH + h) * kS + q) * kS) + kk] = acc[i] + bias;
    }
  }
}

// ---------------------------------------------------------------------------
// Stage 5: attention, one (b,h) per block, fully LDS-resident.
// scores = QK^T/8 + smol; softcap tanh; softmax; out = A @ V -> hcat bf16.
// ---------------------------------------------------------------------------
__global__ __launch_bounds__(256) void k_attention(
    const bf16* __restrict__ Qb, const bf16* __restrict__ Kb,
    const bf16* __restrict__ Vb, const float* __restrict__ smol,
    bf16* __restrict__ hcat) {
  __shared__ __align__(16) bf16 Qs[kS * kDK];
  __shared__ __align__(16) bf16 Ks[kS * kDK];
  __shared__ __align__(16) bf16 Vt[kDK * kS];   // V transposed: [dk][k]
  __shared__ float Sc[kS * kS];
  __shared__ __align__(16) bf16 Abf[kS * kS];

  const int bh = blockIdx.x, b = bh >> 4, h = bh & 15;
  const int tid = threadIdx.x, lane = tid & 31, wave = tid >> 5;
  const size_t base = (size_t)bh * kS * kDK;

  for (int i = tid; i < (kS * kDK) / 8; i += 256) {
    ((v8bf*)Qs)[i] = ((const v8bf*)(Qb + base))[i];
    ((v8bf*)Ks)[i] = ((const v8bf*)(Kb + base))[i];
  }
  for (int i = tid; i < kS * kDK; i += 256) {   // transpose scatter
    const int r = i >> 6, c = i & 63;
    Vt[c * kS + r] = Vb[base + i];
  }
  __syncthreads();

  // ---- scores: 16 tiles / 8 waves -> 2 tiles each
  const int tm = (wave >> 1) * 16;
  #pragma unroll
  for (int t = 0; t < 2; ++t) {
    const int tn = ((wave & 1) * 2 + t) * 16;
    v8f acc = vzero();
    v16bf a0 = load_frag(Qs, tm, kDK, 0,  lane);
    v16bf a1 = load_frag(Qs, tm, kDK, 32, lane);
    v16bf b0 = load_frag(Ks, tn, kDK, 0,  lane);
    v16bf b1 = load_frag(Ks, tn, kDK, 32, lane);
    acc = wmma_bf16(a0, b0, acc);
    acc = wmma_bf16(a1, b1, acc);
    const int colk = tn + (lane & 15);
    #pragma unroll
    for (int i = 0; i < 8; ++i) {
      const int q = tm + i + ((lane >> 4) * 8);
      float v = acc[i] * 0.125f + smol[(size_t)bh * (kS * kS) + q * kS + colk];
      v = tanhf(v * 0.01f) * 100.f;             // soft cap
      Sc[q * kS + colk] = v;
    }
  }
  __syncthreads();

  // ---- softmax per row
  if (tid < kS) {
    float* row = &Sc[tid * kS];
    float m = row[0];
    for (int j = 1; j < kS; ++j) m = fmaxf(m, row[j]);
    float s = 0.f;
    for (int j = 0; j < kS; ++j) { float e = __expf(row[j] - m); row[j] = e; s += e; }
    const float inv = 1.f / s;
    for (int j = 0; j < kS; ++j) Abf[tid * kS + j] = cvt_bf16(row[j] * inv);
  }
  __syncthreads();

  // ---- out = A @ V
  #pragma unroll
  for (int t = 0; t < 2; ++t) {
    const int tn = ((wave & 1) * 2 + t) * 16;   // dk tile
    v8f acc = vzero();
    v16bf a0 = load_frag(Abf, tm, kS, 0,  lane);
    v16bf a1 = load_frag(Abf, tm, kS, 32, lane);
    v16bf b0 = load_frag(Vt, tn, kS, 0,  lane);
    v16bf b1 = load_frag(Vt, tn, kS, 32, lane);
    acc = wmma_bf16(a0, b0, acc);
    acc = wmma_bf16(a1, b1, acc);
    const int dk = tn + (lane & 15);
    #pragma unroll
    for (int i = 0; i < 8; ++i) {
      const int q = tm + i + ((lane >> 4) * 8);
      hcat[(((size_t)b * kS + q) * kD) + h * kDK + dk] = cvt_bf16(acc[i]);
    }
  }
}

// ---------------------------------------------------------------------------
// Launch
// ---------------------------------------------------------------------------
extern "C" void kernel_launch(void* const* d_in, const int* in_sizes, int n_in,
                              void* d_out, int out_size, void* d_ws, size_t ws_size,
                              hipStream_t stream) {
  (void)in_sizes; (void)n_in; (void)out_size; (void)ws_size;
  const float* x      = (const float*)d_in[0];
  const float* query  = (const float*)d_in[1];
  // d_in[2] key_t, d_in[3] value: unused (reference's use_qkv path)
  const float* qkv_w  = (const float*)d_in[4];
  const float* wh_w   = (const float*)d_in[5];
  const float* wh_b   = (const float*)d_in[6];
  const float* sm1_w  = (const float*)d_in[7];
  const float* sm1_b  = (const float*)d_in[8];
  const float* sm2_w  = (const float*)d_in[9];
  const float* sm2_b  = (const float*)d_in[10];
  const float* ln1_g  = (const float*)d_in[11];
  const float* ln1_b  = (const float*)d_in[12];
  const float* sm3_w  = (const float*)d_in[13];
  const float* sm3_b  = (const float*)d_in[14];
  const float* ln2_g  = (const float*)d_in[15];
  const float* ln2_b  = (const float*)d_in[16];
  const float* prep_w = (const float*)d_in[17];
  const float* prep_b = (const float*)d_in[18];
  float* out = (float*)d_out;

  // workspace carve-up (256B aligned)
  char* cur = (char*)d_ws;
  auto alloc = [&](size_t bytes) -> void* {
    void* p = cur; cur += (bytes + 255) & ~(size_t)255; return p;
  };
  const size_t nAct = (size_t)kB * kS * kD;            // 64Mi elems
  bf16* q_bf    = (bf16*)alloc(nAct * 2);
  bf16* x_bf    = (bf16*)alloc(nAct * 2);
  bf16* qkvw_bf = (bf16*)alloc((size_t)3 * kD * kD * 2);
  bf16* whw_bf  = (bf16*)alloc((size_t)kD * kD * 2);
  bf16* sm1w_bf = (bf16*)alloc((size_t)kSMSQ * kD * 2);
  bf16* prepw_bf= (bf16*)alloc((size_t)kS * kS * kSMI * 2);
  bf16* Qb      = (bf16*)alloc((size_t)kB * kH * kS * kDK * 2);
  bf16* Kb      = (bf16*)alloc((size_t)kB * kH * kS * kDK * 2);
  bf16* Vb      = (bf16*)alloc((size_t)kB * kH * kS * kDK * 2);
  float* s1     = (float*)alloc((size_t)kB * kS * kSMSQ * 4);
  bf16* s3bf    = (bf16*)alloc((size_t)kB * kH * kSMI * 2);
  float* smol   = (float*)alloc((size_t)kB * kH * kS * kS * 4);
  bf16* hcat    = (bf16*)alloc(nAct * 2);

  auto pack = [&](const float* src, bf16* dst, size_t n) {
    long n4 = (long)(n / 4);
    k_pack_bf16<<<(unsigned)((n4 + 255) / 256), 256, 0, stream>>>(src, dst, n4);
  };
  pack(query, q_bf, nAct);
  pack(x,     x_bf, nAct);
  pack(qkv_w, qkvw_bf, (size_t)3 * kD * kD);
  pack(wh_w,  whw_bf,  (size_t)kD * kD);
  pack(sm1_w, sm1w_bf, (size_t)kSMSQ * kD);
  pack(prep_w, prepw_bf, (size_t)kS * kS * kSMI);

  k_qkv_gemm<<<dim3(kB, 12), 256, 0, stream>>>(q_bf, qkvw_bf, Qb, Kb, Vb);
  k_sm1_gemm<<<kB, 256, 0, stream>>>(x_bf, sm1w_bf, sm1_b, s1);
  k_smolgen_mlp<<<kB, 256, 0, stream>>>(s1, sm2_w, sm2_b, ln1_g, ln1_b,
                                        sm3_w, sm3_b, ln2_g, ln2_b, s3bf);
  k_smol_prep<<<kB, 256, 0, stream>>>(s3bf, prepw_bf, prep_b, smol);
  k_attention<<<kB * kH, 256, 0, stream>>>(Qb, Kb, Vb, smol, hcat);
  k_out_gemm<<<dim3(kB, 4), 256, 0, stream>>>(hcat, whw_bf, wh_b, out);
}
